// KANOriginal_53687091200525
// MI455X (gfx1250) — compile-verified, matrix operand order checked by reading, not run on hardware
//
#include <hip/hip_runtime.h>
#include <hip/hip_bf16.h>

typedef __attribute__((ext_vector_type(16))) __bf16 v16bf;
typedef __attribute__((ext_vector_type(8)))  float  v8f;

#define IN_F      512
#define OUT_F     512
#define KTOT      3584            // IN_F * 7 (silu + 6 spline bases per input feature)
#define KTILES    112             // KTOT / 32
#define TILE_STRIDE (KTILES * 32 * 16)  // bf16 elems between swizzled 16-col output tiles
#define LDS_STRIDE 3592           // KTOT + 8 bf16 pad -> row offset 7184B = 1796 dwords (bank-spread)
#define LDS_BYTES  (16 * LDS_STRIDE * 2)

// Cubic B-spline (K=3) on the uniform extended grid t_j = -3 + j*(2/3), j=0..9 -> 6 basis fns.
__device__ __forceinline__ void bspline6(float x, float bs[6]) {
    const float h = 2.0f / 3.0f;
    float B[9];
#pragma unroll
    for (int j = 0; j < 9; ++j) {
        float tj = -3.0f + h * (float)j;
        B[j] = (x >= tj && x < tj + h) ? 1.0f : 0.0f;
    }
#pragma unroll
    for (int p = 1; p <= 3; ++p) {
        float inv = 1.0f / (h * (float)p);
#pragma unroll
        for (int j = 0; j < 9 - p; ++j) {
            float tj    = -3.0f + h * (float)j;
            float left  = (x - tj) * inv;
            float right = ((tj + h * (float)(p + 1)) - x) * inv;
            B[j] = left * B[j] + right * B[j + 1];
        }
    }
#pragma unroll
    for (int j = 0; j < 6; ++j) bs[j] = B[j];
}

// Build fused weight W[o][k], k = i*7 + b, pre-swizzled into WMMA-B fragment order:
// fragment (ot,kt): lane l holds col n=l&15, khalf=l>>4; element e<8 -> K=khalf*8+e,
// e>=8 -> K=16+khalf*8+(e-8). 16 bf16 (32B) contiguous per lane, lanes contiguous.
__global__ void __launch_bounds__(256) kan_prep_w(const float* __restrict__ coef,
                                                  const float* __restrict__ scale_base,
                                                  const float* __restrict__ scale_sp,
                                                  __bf16* __restrict__ wsw) {
    const int tid  = blockIdx.x * 256 + threadIdx.x;    // 0 .. 114687
    const int frag = tid >> 5;                          // (ot*KTILES + kt)
    const int lane = tid & 31;
    const int ot    = frag / KTILES;
    const int kt    = frag % KTILES;
    const int n     = lane & 15;
    const int khalf = lane >> 4;
    const int o     = ot * 16 + n;

    __bf16 vals[16];
#pragma unroll
    for (int e = 0; e < 16; ++e) {
        int klocal = (e < 8) ? (khalf * 8 + e) : (16 + khalf * 8 + (e - 8));
        int k = kt * 32 + klocal;
        int i = k / 7;
        int b = k % 7;
        float v;
        if (b == 0) v = scale_base[o * IN_F + i];
        else        v = scale_sp[o * IN_F + i] * coef[(size_t)(o * IN_F + i) * 6 + (b - 1)];
        vals[e] = (__bf16)v;
    }
    uint4* dst = (uint4*)(wsw + (size_t)tid * 16);
    dst[0] = ((const uint4*)vals)[0];
    dst[1] = ((const uint4*)vals)[1];
}

// One workgroup per 16-token block: build bf16 activation panel A[16][KTOT] in LDS,
// then 8 waves, each holding 4 output tiles live: 1 A-fragment feeds 4 WMMAs.
__global__ void __launch_bounds__(256) kan_gemm(const float* __restrict__ x,
                                                const __bf16* __restrict__ wsw,
                                                const float* __restrict__ bias,
                                                float* __restrict__ out) {
    extern __shared__ __align__(16) __bf16 As[];
    const int t0 = blockIdx.x * 16;

    // ---- Phase 1: activations (silu + 6 spline bases) -> LDS, k = feat*7 + b ----
#pragma unroll 4
    for (int it = 0; it < 32; ++it) {
        int eidx = threadIdx.x + it * 256;      // 0..8191 = 16 tokens * 512 feats
        int tok  = eidx >> 9;
        int feat = eidx & 511;
        float xv  = x[(size_t)(t0 + tok) * IN_F + feat];
        float sig = 1.0f / (1.0f + __expf(-xv));
        __bf16* dst = As + tok * LDS_STRIDE + feat * 7;
        dst[0] = (__bf16)(xv * sig);
        float bs[6];
        bspline6(xv, bs);
#pragma unroll
        for (int b = 0; b < 6; ++b) dst[1 + b] = (__bf16)bs[b];
    }
    __syncthreads();

    // ---- Phase 2: WMMA GEMM, 4 output tiles per wave held in registers ----
    const int wave  = threadIdx.x >> 5;
    const int lane  = threadIdx.x & 31;
    const int row   = lane & 15;
    const int khalf = lane >> 4;
    // A fragment base for this lane: row M=lane&15, K-half select by lane>>4
    const __bf16* arow = As + row * LDS_STRIDE + khalf * 8;
    // B base: tile ot = wave*4 + tt at immediate offset tt*TILE_STRIDE (fits 24-bit IOFFSET)
    const __bf16* wb = wsw + ((size_t)(wave * 4) * KTILES * 32 + lane) * 16;

    v8f c0 = {}, c1 = {}, c2 = {}, c3 = {};
#pragma unroll 2
    for (int kt = 0; kt < KTILES; ++kt) {
        v16bf a, b0, b1, b2, b3;
        const uint4* ap = (const uint4*)(arow + kt * 32);
        ((uint4*)&a)[0] = ap[0];        // K = khalf*8 + 0..7
        ((uint4*)&a)[1] = ap[2];        // K = 16 + khalf*8 + 0..7
        const uint4* bp = (const uint4*)(wb + (size_t)kt * 512);
        ((uint4*)&b0)[0] = bp[0];
        ((uint4*)&b0)[1] = bp[1];
        ((uint4*)&b1)[0] = bp[TILE_STRIDE / 8 + 0];
        ((uint4*)&b1)[1] = bp[TILE_STRIDE / 8 + 1];
        ((uint4*)&b2)[0] = bp[2 * (TILE_STRIDE / 8) + 0];
        ((uint4*)&b2)[1] = bp[2 * (TILE_STRIDE / 8) + 1];
        ((uint4*)&b3)[0] = bp[3 * (TILE_STRIDE / 8) + 0];
        ((uint4*)&b3)[1] = bp[3 * (TILE_STRIDE / 8) + 1];
        c0 = __builtin_amdgcn_wmma_f32_16x16x32_bf16(false, a, false, b0, (short)0, c0, false, false);
        c1 = __builtin_amdgcn_wmma_f32_16x16x32_bf16(false, a, false, b1, (short)0, c1, false, false);
        c2 = __builtin_amdgcn_wmma_f32_16x16x32_bf16(false, a, false, b2, (short)0, c2, false, false);
        c3 = __builtin_amdgcn_wmma_f32_16x16x32_bf16(false, a, false, b3, (short)0, c3, false, false);
    }

    // C/D layout: lane n = col (lane&15); rows m = (lane>=16 ? 8 : 0) + r
    const int m0 = t0 + khalf * 8;
    const int n  = lane & 15;
    v8f acc[4] = {c0, c1, c2, c3};
#pragma unroll
    for (int tt = 0; tt < 4; ++tt) {
        const int o   = (wave * 4 + tt) * 16 + n;
        const float bb = bias[o];
#pragma unroll
        for (int r = 0; r < 8; ++r)
            out[(size_t)(m0 + r) * OUT_F + o] = acc[tt][r] + bb;
    }
}

extern "C" void kernel_launch(void* const* d_in, const int* in_sizes, int n_in,
                              void* d_out, int out_size, void* d_ws, size_t ws_size,
                              hipStream_t stream) {
    const float* x          = (const float*)d_in[0];
    const float* coef       = (const float*)d_in[1];
    const float* scale_base = (const float*)d_in[2];
    const float* scale_sp   = (const float*)d_in[3];
    const float* bias       = (const float*)d_in[4];
    __bf16* wsw = (__bf16*)d_ws;                 // OUT_F * KTOT bf16 = 3,670,016 B

    // 32 out-tiles * 112 k-tiles fragments, 32 lanes each -> 114688 threads
    kan_prep_w<<<448, 256, 0, stream>>>(coef, scale_base, scale_sp, wsw);
    // 16384 tokens / 16 per block
    kan_gemm<<<1024, 256, LDS_BYTES, stream>>>(x, wsw, bias, (float*)d_out);
}
